// GCEncoderLayer_78400333021790
// MI455X (gfx1250) — matrix operand order repaired and checked by reading, not compiled, not run on hardware
//
#include <hip/hip_runtime.h>

#define N_SENS   325
#define T_SEQ    12
#define DM       128
#define NH       8
#define KD       128
#define DFF_     512
#define NNZ_     3250
#define BATCH    32
#define ROWS     124800   /* 32*325*12 */
#define BT_      384      /* 32*12 */

typedef __attribute__((ext_vector_type(16))) __bf16 v16bf;
typedef __attribute__((ext_vector_type(8)))  __bf16 v8bf;
typedef __attribute__((ext_vector_type(8)))  float  v8f;

__device__ __forceinline__ __bf16 f2bf(float f) {
  unsigned u = __builtin_bit_cast(unsigned, f);
  unsigned short r = (unsigned short)((u + 0x7FFFu + ((u >> 16) & 1u)) >> 16);
  return __builtin_bit_cast(__bf16, r);
}

__device__ __forceinline__ v8f zero8() {
  v8f z = {0.f, 0.f, 0.f, 0.f, 0.f, 0.f, 0.f, 0.f};
  return z;
}

// A fragment (16x32 bf16) from LDS: lane half h holds K = [16h,16h+16)
__device__ __forceinline__ v16bf frag_a(const __bf16* base, int ld, int lane) {
  int m = lane & 15, h = lane >> 4;
  const __bf16* rp = base + m * ld;
  v8bf lo = *(const v8bf*)(rp + h * 16);
  v8bf hi = *(const v8bf*)(rp + h * 16 + 8);
  return __builtin_shufflevector(lo, hi, 0,1,2,3,4,5,6,7,8,9,10,11,12,13,14,15);
}

// B fragment (32x16 bf16) from a transposed LDS tile Bt[n][k]: lane half h holds K = [16h,16h+16)
__device__ __forceinline__ v16bf frag_b(const __bf16* base, int ld, int lane) {
  int n = lane & 15, h = lane >> 4;
  const __bf16* rp = base + n * ld + h * 16;
  v8bf lo = *(const v8bf*)(rp);
  v8bf hi = *(const v8bf*)(rp + 8);
  return __builtin_shufflevector(lo, hi, 0,1,2,3,4,5,6,7,8,9,10,11,12,13,14,15);
}

// B fragment directly from fragment-ordered packed weights in global memory:
// lane's 16 bf16 live at wt[(tile*32 + lane)*16 .. +16)  (one coalesced 32B load)
__device__ __forceinline__ v16bf frag_b_w(const __bf16* wt, int tile, int lane) {
  const __bf16* rp = wt + (((size_t)tile * 32 + lane) << 4);
  v8bf lo = *(const v8bf*)(rp);
  v8bf hi = *(const v8bf*)(rp + 8);
  return __builtin_shufflevector(lo, hi, 0,1,2,3,4,5,6,7,8,9,10,11,12,13,14,15);
}

__device__ __forceinline__ v8f wmma_bf(v16bf a, v16bf b, v8f c) {
  return __builtin_amdgcn_wmma_f32_16x16x32_bf16(false, a, false, b, (short)0, c, false, false);
}

// map flat packed index -> (k, n) of the original weight; KT = K/32, tiles are (nt, kt) row-major
__device__ __forceinline__ void tilepos(int i, int KT, int& kg, int& ng) {
  int tile = i >> 9, r = i & 511;
  int lane = r >> 4, e = r & 15;
  int nt = tile / KT, kt = tile - nt * KT;
  kg = kt * 32 + (lane >> 4) * 16 + e;
  ng = nt * 16 + (lane & 15);
}

// ---------------------------------------------------------------- K0: pack weights to bf16, fragment order
__global__ __launch_bounds__(256) void pack_weights_kernel(
    const float* __restrict__ Wq, const float* __restrict__ Wk, const float* __restrict__ Wv,
    const float* __restrict__ Wo, const float* __restrict__ W1, const float* __restrict__ W2,
    const float* __restrict__ gc,
    __bf16* __restrict__ WqkvP, __bf16* __restrict__ WoP, __bf16* __restrict__ W1P,
    __bf16* __restrict__ W2P, __bf16* __restrict__ gcP) {
  int i = blockIdx.x * 256 + threadIdx.x;
  int kg, ng;
  if (i < 3072 * 128) {  // N=3072 (q|k|v * 1024 + h*128 + kd), K=128 (d), KT=4
    tilepos(i, 4, kg, ng);
    float v = (ng < 1024) ? Wq[kg * 1024 + ng]
            : (ng < 2048) ? Wk[kg * 1024 + (ng - 1024)]
                          : Wv[kg * 1024 + (ng - 2048)];
    WqkvP[i] = f2bf(v);
  }
  if (i < 128 * 1024) {  // N=128 (dout), K=1024 (hk), KT=32
    tilepos(i, 32, kg, ng);
    WoP[i] = f2bf(Wo[kg * 128 + ng]);
  }
  if (i < 512 * 128) {   // N=512, K=128, KT=4
    tilepos(i, 4, kg, ng);
    W1P[i] = f2bf(W1[kg * 512 + ng]);
  }
  if (i < 128 * 512) {   // N=128, K=512, KT=16
    tilepos(i, 16, kg, ng);
    W2P[i] = f2bf(W2[kg * 128 + ng]);
  }
  if (i < 128 * 384) {   // N=128, K=384, KT=12
    tilepos(i, 12, kg, ng);
    gcP[i] = f2bf(gc[kg * 128 + ng]);
  }
}

// ---------------------------------------------------------------- K1: fused MHA + residual + LN1
__global__ __launch_bounds__(256) void attn_ln1_kernel(
    const float* __restrict__ x, const __bf16* __restrict__ WqkvP,
    const __bf16* __restrict__ WoP,
    const float* __restrict__ bq, const float* __restrict__ bk,
    const float* __restrict__ bv, const float* __restrict__ bo,
    const float* __restrict__ g1, const float* __restrict__ be1,
    float* __restrict__ x1) {
  __shared__ __attribute__((aligned(16))) __bf16 Xs[16 * 128];
  __shared__ __attribute__((aligned(16))) float  Xf[12 * 128];
  __shared__ __attribute__((aligned(16))) __bf16 Qh[16 * 128];
  __shared__ __attribute__((aligned(16))) __bf16 Kh[16 * 128];
  __shared__ __attribute__((aligned(16))) __bf16 Vt[128 * 32];
  __shared__ __attribute__((aligned(16))) float  Sc[16 * 16];
  __shared__ __attribute__((aligned(16))) __bf16 At[16 * 32];
  __shared__ __attribute__((aligned(16))) __bf16 Cx[16 * 128];
  __shared__ __attribute__((aligned(16))) float  Oacc[16 * 128];
  __shared__ float mu[16], rs[16];

  const int bn = blockIdx.x;
  const int tid = (int)threadIdx.x;
  const int w = tid >> 5;
  const int lane = tid & 31;
  const float* xrow = x + (size_t)bn * (T_SEQ * DM);

  // stage input tile (pad rows 12..15 with zeros); zero Vt / At padding regions once
  for (int i = tid; i < 16 * 128; i += 256) {
    int m = i >> 7, d = i & 127;
    float v = (m < 12) ? xrow[m * 128 + d] : 0.f;
    if (m < 12) Xf[m * 128 + d] = v;
    Xs[i] = f2bf(v);
  }
  for (int i = tid; i < 128 * 32; i += 256) Vt[i] = f2bf(0.f);
  for (int i = tid; i < 16 * 32; i += 256) At[i] = f2bf(0.f);

  v8f oacc = zero8();  // persistent Wo-projection accumulator: wave w owns out cols [16w,16w+16)

  for (int h = 0; h < NH; ++h) {
    __syncthreads();
    // ---- QKV for head h: 24 n-tiles across 8 waves; B fragments straight from packed weights
    for (int it = 0; it < 3; ++it) {
      int tile = w + it * 8;
      int which = tile >> 3;        // 0=q 1=k 2=v
      int nt = tile & 7;
      int ncol = which * 1024 + h * 128 + nt * 16;
      v8f acc = zero8();
      #pragma unroll
      for (int kk = 0; kk < 4; ++kk) {
        v16bf a = frag_a(Xs + kk * 32, 128, lane);
        v16bf b = frag_b_w(WqkvP, (ncol >> 4) * 4 + kk, lane);
        acc = wmma_bf(a, b, acc);
      }
      const float* bias = (which == 0) ? bq : (which == 1) ? bk : bv;
      int n = lane & 15, mh = lane >> 4;
      #pragma unroll
      for (int r = 0; r < 8; ++r) {
        int m = r + 8 * mh;
        __bf16 bb = f2bf(acc[r] + bias[h * 128 + nt * 16 + n]);
        if (which == 0)      Qh[m * 128 + nt * 16 + n] = bb;
        else if (which == 1) Kh[m * 128 + nt * 16 + n] = bb;
        else                 Vt[(nt * 16 + n) * 32 + m] = bb;  // V transposed [kd][s]
      }
    }
    __syncthreads();
    // ---- scores = Q K^T / sqrt(128)  (wave 0)
    if (w == 0) {
      v8f acc = zero8();
      #pragma unroll
      for (int kk = 0; kk < 4; ++kk) {
        v16bf a = frag_a(Qh + kk * 32, 128, lane);
        v16bf b = frag_b(Kh + kk * 32, 128, lane);
        acc = wmma_bf(a, b, acc);
      }
      int n = lane & 15, mh = lane >> 4;
      #pragma unroll
      for (int r = 0; r < 8; ++r) Sc[(r + 8 * mh) * 16 + n] = acc[r] * 0.08838834764831845f;
    }
    __syncthreads();
    // ---- softmax over the 12 valid key positions
    if (tid < 12) {
      float mx = -1e30f;
      #pragma unroll
      for (int s = 0; s < 12; ++s) mx = fmaxf(mx, Sc[tid * 16 + s]);
      float e[12], sum = 0.f;
      #pragma unroll
      for (int s = 0; s < 12; ++s) { e[s] = __expf(Sc[tid * 16 + s] - mx); sum += e[s]; }
      float inv = 1.f / sum;
      #pragma unroll
      for (int s = 0; s < 12; ++s) At[tid * 32 + s] = f2bf(e[s] * inv);
    }
    __syncthreads();
    // ---- ctx = attn @ V : wave w computes kd cols [16w,16w+16)
    {
      v8f acc = zero8();
      v16bf a = frag_a(At, 32, lane);
      v16bf b = frag_b(Vt + (w * 16) * 32, 32, lane);
      acc = wmma_bf(a, b, acc);
      int n = lane & 15, mh = lane >> 4;
      #pragma unroll
      for (int r = 0; r < 8; ++r) Cx[(r + 8 * mh) * 128 + w * 16 + n] = f2bf(acc[r]);
    }
    __syncthreads();
    // ---- out += ctx_h @ Wo[h] : accumulate into persistent registers
    #pragma unroll
    for (int kk = 0; kk < 4; ++kk) {
      v16bf a = frag_a(Cx + kk * 32, 128, lane);
      v16bf b = frag_b_w(WoP, w * 32 + h * 4 + kk, lane);  // nt=w, KT=32, kt=h*4+kk
      oacc = wmma_bf(a, b, oacc);
    }
  }
  // ---- epilogue: + bo + residual, LayerNorm, store
  {
    int n = lane & 15, mh = lane >> 4;
    #pragma unroll
    for (int r = 0; r < 8; ++r) {
      int m = r + 8 * mh, d = w * 16 + n;
      float v = oacc[r] + bo[d];
      Oacc[m * 128 + d] = (m < 12) ? (v + Xf[m * 128 + d]) : 0.f;
    }
  }
  __syncthreads();
  if (tid < 12) {
    float s = 0.f, s2 = 0.f;
    for (int d = 0; d < 128; ++d) { float v = Oacc[tid * 128 + d]; s += v; s2 += v * v; }
    float m_ = s * (1.f / 128.f);
    mu[tid] = m_;
    rs[tid] = rsqrtf(s2 * (1.f / 128.f) - m_ * m_ + 1e-3f);
  }
  __syncthreads();
  for (int i = tid; i < 12 * 128; i += 256) {
    int m = i >> 7, d = i & 127;
    x1[(size_t)bn * (12 * 128) + i] = (Oacc[i] - mu[m]) * rs[m] * g1[d] + be1[d];
  }
}

// ---------------------------------------------------------------- K2: fused FFN + residual + LN2
__global__ __launch_bounds__(256) void ffn_ln2_kernel(
    const float* __restrict__ x1, const __bf16* __restrict__ W1P, const float* __restrict__ b1,
    const __bf16* __restrict__ W2P, const float* __restrict__ b2,
    const float* __restrict__ g2, const float* __restrict__ be2,
    float* __restrict__ x2) {
  __shared__ __attribute__((aligned(16))) __bf16 Xs[16 * 128];
  __shared__ __attribute__((aligned(16))) float  Xf[16 * 128];
  __shared__ __attribute__((aligned(16))) __bf16 Hs[16 * 512];
  __shared__ __attribute__((aligned(16))) float  Oacc[16 * 128];
  __shared__ float mu[16], rs[16];

  const int tid = (int)threadIdx.x;
  const int w = tid >> 5, lane = tid & 31;
  const size_t row0 = (size_t)blockIdx.x * 16;
  const float* src = x1 + row0 * 128;

  for (int i = tid; i < 16 * 128; i += 256) {
    float v = src[i];
    Xf[i] = v;
    Xs[i] = f2bf(v);
  }
  __syncthreads();
  // h = relu(x1 @ W1 + b1) -> LDS bf16
  for (int j = 0; j < 4; ++j) {
    int nt = w + j * 8;
    int ncol = nt * 16;
    v8f acc = zero8();
    #pragma unroll
    for (int kk = 0; kk < 4; ++kk) {
      v16bf a = frag_a(Xs + kk * 32, 128, lane);
      v16bf b = frag_b_w(W1P, nt * 4 + kk, lane);
      acc = wmma_bf(a, b, acc);
    }
    int n = lane & 15, mh = lane >> 4;
    #pragma unroll
    for (int r = 0; r < 8; ++r)
      Hs[(r + 8 * mh) * 512 + ncol + n] = f2bf(fmaxf(acc[r] + b1[ncol + n], 0.f));
  }
  __syncthreads();
  // out = h @ W2 + b2 + residual
  {
    v8f acc = zero8();
    for (int kk = 0; kk < 16; ++kk) {
      v16bf a = frag_a(Hs + kk * 32, 512, lane);
      v16bf b = frag_b_w(W2P, w * 16 + kk, lane);
      acc = wmma_bf(a, b, acc);
    }
    int n = lane & 15, mh = lane >> 4;
    #pragma unroll
    for (int r = 0; r < 8; ++r) {
      int m = r + 8 * mh, d = w * 16 + n;
      Oacc[m * 128 + d] = acc[r] + b2[d] + Xf[m * 128 + d];
    }
  }
  __syncthreads();
  if (tid < 16) {
    float s = 0.f, s2 = 0.f;
    for (int d = 0; d < 128; ++d) { float v = Oacc[tid * 128 + d]; s += v; s2 += v * v; }
    float m_ = s * (1.f / 128.f);
    mu[tid] = m_;
    rs[tid] = rsqrtf(s2 * (1.f / 128.f) - m_ * m_ + 1e-3f);
  }
  __syncthreads();
  for (int i = tid; i < 16 * 128; i += 256) {
    int m = i >> 7, d = i & 127;
    x2[row0 * 128 + i] = (Oacc[i] - mu[m]) * rs[m] * g2[d] + be2[d];
  }
}

// ---------------------------------------------------------------- K3: zero scratch
__global__ __launch_bounds__(256) void zero_f32_kernel(float* __restrict__ p, long long n) {
  for (long long i = (long long)blockIdx.x * 256 + threadIdx.x; i < n;
       i += (long long)gridDim.x * 256)
    p[i] = 0.f;
}

// ---------------------------------------------------------------- K4: COO SpMM via f32 atomics
__global__ __launch_bounds__(256) void spmm_kernel(
    const float* __restrict__ x2,
    const float* __restrict__ v0, const int* __restrict__ r0, const int* __restrict__ c0,
    const float* __restrict__ v1, const int* __restrict__ r1, const int* __restrict__ c1,
    float* __restrict__ s1, float* __restrict__ s2) {
  const int e = blockIdx.x, b = blockIdx.y, sel = blockIdx.z;
  const float val = sel ? v1[e] : v0[e];
  const int row = sel ? r1[e] : r0[e];
  const int col = sel ? c1[e] : c0[e];
  float* sout = sel ? s2 : s1;
  const float* src = x2 + ((size_t)(b * N_SENS + col) * T_SEQ) * DM;
  for (int i = (int)threadIdx.x; i < T_SEQ * DM; i += 256) {
    int t = i >> 7, d = i & 127;
    atomicAdd(&sout[((size_t)((b * T_SEQ + t) * N_SENS + row)) * DM + d], val * src[t * DM + d]);
  }
}

// ---------------------------------------------------------------- K5: graph-conv GEMM (K=384 interleaved) + output permute
__global__ __launch_bounds__(256) void gc_gemm_kernel(
    const float* __restrict__ x2, const float* __restrict__ s1, const float* __restrict__ s2,
    const __bf16* __restrict__ gcP, const float* __restrict__ gbias,
    float* __restrict__ out) {
  __shared__ __attribute__((aligned(16))) __bf16 As[16 * 384];
  const int tid = (int)threadIdx.x;
  const int w = tid >> 5, lane = tid & 31;
  const int row0 = blockIdx.x * 16;  // rows in (bt, n_sensor) order

  // stage A tile: feature k = d*3 + support (matches the reference transpose/reshape)
  for (int i = tid; i < 16 * 384; i += 256) {
    int m = i / 384, k = i - m * 384;
    int mm = row0 + m;
    int bt = mm / N_SENS, ns = mm - bt * N_SENS;
    int b = bt / T_SEQ, t = bt - b * T_SEQ;
    int d = k / 3, sel = k - d * 3;
    float v;
    if (sel == 0)      v = x2[((size_t)((b * N_SENS + ns) * T_SEQ + t)) * DM + d];
    else if (sel == 1) v = s1[(size_t)mm * DM + d];
    else               v = s2[(size_t)mm * DM + d];
    As[i] = f2bf(v);
  }
  __syncthreads();
  v8f acc = zero8();
  for (int kk = 0; kk < 12; ++kk) {
    v16bf a = frag_a(As + kk * 32, 384, lane);
    v16bf b = frag_b_w(gcP, w * 12 + kk, lane);
    acc = wmma_bf(a, b, acc);
  }
  int n = lane & 15, mh = lane >> 4;
  #pragma unroll
  for (int r = 0; r < 8; ++r) {
    int m = r + 8 * mh, mm = row0 + m;
    int bt = mm / N_SENS, ns = mm - bt * N_SENS;
    int b = bt / T_SEQ, t = bt - b * T_SEQ;
    out[((size_t)((b * N_SENS + ns) * T_SEQ + t)) * DM + w * 16 + n] = acc[r] + gbias[w * 16 + n];
  }
}

// ----------------------------------------------------------------
extern "C" void kernel_launch(void* const* d_in, const int* in_sizes, int n_in,
                              void* d_out, int out_size, void* d_ws, size_t ws_size,
                              hipStream_t stream) {
  (void)in_sizes; (void)n_in; (void)out_size; (void)ws_size;
  const float* x      = (const float*)d_in[0];
  const float* Wq     = (const float*)d_in[1];
  const float* bq     = (const float*)d_in[2];
  const float* Wk     = (const float*)d_in[3];
  const float* bk     = (const float*)d_in[4];
  const float* Wv     = (const float*)d_in[5];
  const float* bv     = (const float*)d_in[6];
  const float* Wo     = (const float*)d_in[7];
  const float* bo     = (const float*)d_in[8];
  const float* ln1_g  = (const float*)d_in[9];
  const float* ln1_b  = (const float*)d_in[10];
  const float* fW1    = (const float*)d_in[11];
  const float* fb1    = (const float*)d_in[12];
  const float* fW2    = (const float*)d_in[13];
  const float* fb2    = (const float*)d_in[14];
  const float* ln2_g  = (const float*)d_in[15];
  const float* ln2_b  = (const float*)d_in[16];
  const float* gck    = (const float*)d_in[17];
  const float* gcb    = (const float*)d_in[18];
  const float* s0v    = (const float*)d_in[19];
  const float* s1v    = (const float*)d_in[20];
  const int*   s0r    = (const int*)d_in[21];
  const int*   s0c    = (const int*)d_in[22];
  const int*   s1r    = (const int*)d_in[23];
  const int*   s1c    = (const int*)d_in[24];

  char* ws = (char*)d_ws;
  size_t off = 0;
  auto take = [&](size_t bytes) -> void* {
    void* p = ws + off;
    off += (bytes + 255) & ~(size_t)255;
    return p;
  };
  __bf16* WqkvP = (__bf16*)take((size_t)3072 * 128 * 2);
  __bf16* WoP   = (__bf16*)take((size_t)128 * 1024 * 2);
  __bf16* W1P   = (__bf16*)take((size_t)512 * 128 * 2);
  __bf16* W2P   = (__bf16*)take((size_t)128 * 512 * 2);
  __bf16* gcP   = (__bf16*)take((size_t)128 * 384 * 2);
  float*  x1    = (float*)take((size_t)ROWS * DM * 4);
  float*  x2    = (float*)take((size_t)ROWS * DM * 4);
  float*  sA    = (float*)take((size_t)2 * ROWS * DM * 4);  // s1 | s2 contiguous
  float*  sB    = sA + (size_t)ROWS * DM;

  pack_weights_kernel<<<dim3((3072 * 128 + 255) / 256), dim3(256), 0, stream>>>(
      Wq, Wk, Wv, Wo, fW1, fW2, gck, WqkvP, WoP, W1P, W2P, gcP);
  attn_ln1_kernel<<<dim3(BATCH * N_SENS), dim3(256), 0, stream>>>(
      x, WqkvP, WoP, bq, bk, bv, bo, ln1_g, ln1_b, x1);
  ffn_ln2_kernel<<<dim3(ROWS / 16), dim3(256), 0, stream>>>(
      x1, W1P, fb1, W2P, fb2, ln2_g, ln2_b, x2);
  zero_f32_kernel<<<dim3(4096), dim3(256), 0, stream>>>(sA, (long long)2 * ROWS * DM);
  spmm_kernel<<<dim3(NNZ_, BATCH, 2), dim3(256), 0, stream>>>(
      x2, s0v, s0r, s0c, s1v, s1r, s1c, sA, sB);
  gc_gemm_kernel<<<dim3(ROWS / 16), dim3(256), 0, stream>>>(
      x2, sA, sB, gcP, gcb, (float*)d_out);
}
